// ConvLIF_WTA_17059610100041
// MI455X (gfx1250) — matrix operand order repaired and checked by reading, not compiled, not run on hardware
//
#include <hip/hip_runtime.h>

typedef __attribute__((ext_vector_type(16))) _Float16     v16h;
typedef __attribute__((ext_vector_type(8)))  float        v8f;
typedef __attribute__((ext_vector_type(4)))  unsigned int v4u;
typedef __attribute__((ext_vector_type(8)))  int          v8i;
typedef __attribute__((ext_vector_type(4)))  int          v4i;

#define T_LEN  4096
#define K_CH   64
#define KS     16
#define TCHUNK 64
#define SPAD   68              // slds row stride (floats): 16B-aligned, de-conflicted

#define ALPHA_C 0.9048374180359595732f   // exp(-1/10)
#define BETA_C  0.0951625819640404268f   // 1 - ALPHA
#define THETA_C 0.5f

// max with a DPP16 lane-permute of the source (VALU-latency, no LDS round trip)
#define DPP_MAX(v, ctrl)                                                      \
  do {                                                                        \
    int _t = __builtin_amdgcn_mov_dpp(__float_as_int(v), (ctrl), 0xF, 0xF,    \
                                      true);                                  \
    (v) = fmaxf((v), __int_as_float(_t));                                     \
  } while (0)

__global__ __launch_bounds__(32) void convlif_wta_kernel(
    const float* __restrict__ x, const float* __restrict__ W,
    float* __restrict__ out)
{
  __shared__ __align__(16) float xlds[KS + T_LEN];      // 16-float zero pad + x row
  __shared__ __align__(16) float ulds[TCHUNK * K_CH];   // conv chunk, [tt][k]
  __shared__ __align__(16) float slds[K_CH * SPAD];     // spike chunk, [k][tt] padded

  const int lane = threadIdx.x;              // wave32
  const int b    = blockIdx.x;               // one batch row per wave
  const int n    = lane & 15;
  const int half = lane >> 4;

  // causal left-pad x[-15..-1]
  if (lane < KS) xlds[lane] = 0.0f;

  // ---- TDM: async DMA of the x row (4096 f32 = 16KB) into LDS ----------
  const float*       xrow = x + (size_t)b * T_LEN;
  unsigned long long ga   = (unsigned long long)(uintptr_t)xrow;
  unsigned int       la   = (unsigned int)(uintptr_t)(&xlds[KS]);
  v4u g0;
  g0[0] = 1u;                                         // count=1, user mode
  g0[1] = la;                                         // lds_addr (bytes)
  g0[2] = (unsigned int)(ga & 0xFFFFFFFFu);           // global_addr[31:0]
  g0[3] = (unsigned int)((ga >> 32) & 0x1FFFFFFu)     // global_addr[56:32]
        | (2u << 30);                                 // type = 2 ("image")
  v8i g1 = {0,0,0,0,0,0,0,0};
  g1[0] = (int)(2u << 16);                            // data_size = 2 -> 4 bytes
  g1[1] = (int)((4096u & 0xFFFFu) << 16);             // tensor_dim0[15:0] @ bits63:48
  g1[2] = (int)(1u << 16);                            // tensor_dim1 lo = 1
  g1[3] = (int)(4096u << 16);                         // tile_dim0 = 4096
  g1[4] = (int)1u;                                    // tile_dim1 = 1
  g1[5] = (int)4096u;                                 // tensor_dim0_stride[31:0]
  v4i g2 = {0,0,0,0};
  v4i g3 = {0,0,0,0};
  v8i g4 = {0,0,0,0,0,0,0,0};
  __builtin_amdgcn_tensor_load_to_lds(g0, g1, g2, g3, g4, 0);

  // ---- Preload B operands: B[j,k] = W[k,j], 4 k-tiles of 16 channels ----
  v16h Bt[4];
#pragma unroll
  for (int kt = 0; kt < 4; ++kt) {
#pragma unroll
    for (int j = 0; j < 16; ++j) {
      float w = (j < 8) ? W[(kt * 16 + n) * KS + half * 8 + j] : 0.0f;
      Bt[kt][j] = (_Float16)w;
    }
  }

  __builtin_amdgcn_s_wait_tensorcnt(0);   // x row resident in LDS

  const v8f czero = {0.f,0.f,0.f,0.f,0.f,0.f,0.f,0.f};
  float v0 = 0.0f, v1 = 0.0f;             // LIF state: channels 2*lane, 2*lane+1

  for (int t0 = 0; t0 < T_LEN; t0 += TCHUNK) {
    // ---- conv for 64 steps: 4 time-tiles x 4 k-tiles = 16 WMMAs ---------
#pragma unroll
    for (int tt4 = 0; tt4 < 4; ++tt4) {
      const int tb = t0 + tt4 * 16;
      // A[m, j] = x[tb + m - 15 + j] = xlds[tb + m + 1 + j]
      v16h A;
      const int base = tb + 1 + n + half * 8;
#pragma unroll
      for (int j = 0; j < 8; ++j) A[j] = (_Float16)xlds[base + j];
#pragma unroll
      for (int j = 8; j < 16; ++j) A[j] = (_Float16)0.0f;

#pragma unroll
      for (int kt = 0; kt < 4; ++kt) {
        v8f acc = __builtin_amdgcn_wmma_f32_16x16x32_f16(
            false, A, false, Bt[kt], (short)0, czero, false, false);
        // D: VGPR r, lanes 0-15 -> (M=r, N=n); lanes 16-31 -> (M=8+r, N=n)
#pragma unroll
        for (int r = 0; r < 8; ++r)
          ulds[(tt4 * 16 + r + half * 8) * K_CH + kt * 16 + n] = acc[r];
      }
    }

    // ---- 64 sequential LIF + WTA steps (branchless, VALU-only reduce) ---
    for (int tt = 0; tt < TCHUNK; ++tt) {
      const float2 u2 = *(const float2*)&ulds[tt * K_CH + 2 * lane];
      v0 = ALPHA_C * v0 + BETA_C * u2.x;
      v1 = ALPHA_C * v1 + BETA_C * u2.y;

      // all-lane max via DPP butterfly: masks {1,2,7,15,16} generate full
      // 5-bit xor group -> every lane converges to the global max.
      float mv = fmaxf(v0, v1);
      DPP_MAX(mv, 0xB1);    // quad_perm(1,0,3,2)  : xor 1
      DPP_MAX(mv, 0x4E);    // quad_perm(2,3,0,1)  : xor 2
      DPP_MAX(mv, 0x141);   // row_half_mirror     : xor 7
      DPP_MAX(mv, 0x140);   // row_mirror          : xor 15
      {                     // cross-half swap     : xor 16
        int t = __builtin_amdgcn_permlanex16(
            __float_as_int(mv), __float_as_int(mv),
            0x76543210, 0xFEDCBA98, false, false);
        mv = fmaxf(mv, __int_as_float(t));
      }

      // winner index via ballot + ctz (first-index tie-break like argmax)
      unsigned long long b0 = __ballot(v0 == mv);
      unsigned long long b1 = __ballot(v1 == mv);
      int c0 = (int)__builtin_ctzll(b0 | 0x100000000ull);  // 32 if empty
      int c1 = (int)__builtin_ctzll(b1 | 0x100000000ull);
      int wi0 = 2 * c0, wi1 = 2 * c1 + 1;
      int mi  = wi0 < wi1 ? wi0 : wi1;

      float sp = (mv - THETA_C >= 0.0f) ? 1.0f : 0.0f;     // winner spiked?
      float s0 = (mi == 2 * lane)     ? sp : 0.0f;
      float s1 = (mi == 2 * lane + 1) ? sp : 0.0f;
      v0 -= s0 * THETA_C;                                  // winner-only reset
      v1 -= s1 * THETA_C;
      slds[(2 * lane)     * SPAD + tt] = s0;
      slds[(2 * lane + 1) * SPAD + tt] = s1;
    }

    // ---- Flush: each float4-store covers one full 256B row across lanes -
#pragma unroll
    for (int j = 0; j < 32; ++j) {
      const int row  = 2 * j + half;
      const int toff = n * 4;
      const float4 s4 = *(const float4*)&slds[row * SPAD + toff];
      *(float4*)&out[((size_t)b * K_CH + row) * T_LEN + t0 + toff] = s4;
    }
  }
}

extern "C" void kernel_launch(void* const* d_in, const int* in_sizes, int n_in,
                              void* d_out, int out_size, void* d_ws, size_t ws_size,
                              hipStream_t stream) {
  const float* x = (const float*)d_in[0];   // [B,1,T] fp32
  const float* W = (const float*)d_in[1];   // [K,1,16] fp32
  float* out = (float*)d_out;               // [B,K,T] fp32

  const int B = in_sizes[0] / T_LEN;        // 256
  convlif_wta_kernel<<<dim3(B), dim3(32), 0, stream>>>(x, W, out);
}
